// ExplanationGenerator_72438918414598
// MI455X (gfx1250) — matrix operand order repaired and checked by reading, not compiled, not run on hardware
//
#include <hip/hip_runtime.h>

// ---------------------------------------------------------------------------
// MI455X / gfx1250 implementation of the 32-step explanation-generator decoder.
// All big GEMMs run on v_wmma_f32_16x16x32_bf16 (bf16 in / fp32 accumulate).
// Weights are converted to bf16 once (~92MB, resident in the 192MB L2) and the
// encoder K/V projections are hoisted out of the step loop (step-invariant).
// OOB M-rows read from a zeroed dummy row -> branchless WMMA inner loop.
// ---------------------------------------------------------------------------

typedef __attribute__((ext_vector_type(16))) __bf16        bf16x16;
typedef __attribute__((ext_vector_type(8)))  float         f32x8;
typedef __attribute__((ext_vector_type(4)))  unsigned int  u32x4;

union Frag { u32x4 q[2]; bf16x16 v; };

__device__ __forceinline__ float bf2f(unsigned short u) {
  unsigned int x = ((unsigned int)u) << 16;
  return __builtin_bit_cast(float, x);
}
__device__ __forceinline__ unsigned short f2bf(float f) {
  unsigned int u = __builtin_bit_cast(unsigned int, f);
  u += 0x7FFFu + ((u >> 16) & 1u);            // round-to-nearest-even
  return (unsigned short)(u >> 16);
}

// ---------------------------------------------------------------------------
// Generic fp32 -> bf16 conversion (weights / hidden states)
// ---------------------------------------------------------------------------
__global__ void f32_to_bf16_kernel(const float* __restrict__ src,
                                   unsigned short* __restrict__ dst, long n) {
  long i = (long)blockIdx.x * blockDim.x + threadIdx.x;
  long stride = (long)gridDim.x * blockDim.x;
  for (; i < n; i += stride) dst[i] = f2bf(src[i]);
}

__global__ void zero_u16_kernel(unsigned short* __restrict__ dst, int n) {
  int i = blockIdx.x * blockDim.x + threadIdx.x;
  if (i < n) dst[i] = 0;
}

// ---------------------------------------------------------------------------
// bf16 WMMA GEMM:  C[M,N] = A[M,K] * W[N,K]^T + bias  (opt. ReLU)
//   A row m lives at  A + (m / ASq)*Abs + (m % ASq)*lda    (ragged decoder buf)
//   Cf row m lives at Cf + (m / CSq)*Cbs + (m % CSq)*ldcf  (strided logits out)
//   Cb (bf16 copy) is contiguous [M, ldcb].
// One wave computes one 16x16 tile; 8 waves/block share an M-tile row.
// OOB M-rows alias a zeroed dummy row, so the K loop is branchless:
// per 32-wide K step it is exactly 4x global_load_b128 + 1x v_wmma.
// ---------------------------------------------------------------------------
struct GemmP {
  const unsigned short* A; long lda; int ASq; long Abs;
  const unsigned short* W; long ldw;
  const float* bias;
  const unsigned short* zrow;   // zero-filled dummy row (>= K elements)
  float* Cf; long ldcf; int CSq; long Cbs;
  unsigned short* Cb; long ldcb;
  int M, N, K, relu;
};

__global__ void gemm_bf16_wmma(GemmP p) {
  const int wave = threadIdx.x >> 5;
  const int lane = threadIdx.x & 31;
  const int n0 = (blockIdx.x * 8 + wave) * 16;
  const int m0 = blockIdx.y * 16;
  if (n0 >= p.N) return;

  const int mlane = m0 + (lane & 15);
  const unsigned short* arow =
      (mlane < p.M)
          ? p.A + (long)(mlane / p.ASq) * p.Abs + (long)(mlane % p.ASq) * p.lda
          : p.zrow;
  const unsigned short* brow = p.W + (long)(n0 + (lane & 15)) * p.ldw;

  // ISA A layout (16-bit, 16x32): lane<16 -> K 0..7 & 16..23 ; lane>=16 -> +8
  // ISA B layout (16-bit, 32x16): lane<16 -> K 0..7 & 16..23 ; lane>=16 -> +8
  const int kA = (lane >= 16) ? 8 : 0;
  const int kB = (lane >= 16) ? 8 : 0;
  const unsigned short* aptr = arow + kA;
  const unsigned short* bptr = brow + kB;

  f32x8 acc = {0.f, 0.f, 0.f, 0.f, 0.f, 0.f, 0.f, 0.f};

  for (int k0 = 0; k0 < p.K; k0 += 32) {
    Frag fa, fb;
    const u32x4* ap = (const u32x4*)(aptr + k0);    // 16B aligned
    fa.q[0] = ap[0];          // K = base .. base+7
    fa.q[1] = ap[2];          // K = base+16 .. base+23
    const u32x4* bp = (const u32x4*)(bptr + k0);    // 16B aligned
    fb.q[0] = bp[0];          // K = base .. base+7
    fb.q[1] = bp[2];          // K = base+16 .. base+23
    acc = __builtin_amdgcn_wmma_f32_16x16x32_bf16(
        false, fa.v, false, fb.v, (short)0, acc, false, false);
  }

  const int col = n0 + (lane & 15);
  const float bv = p.bias ? p.bias[col] : 0.f;
  const int rbase = m0 + ((lane >> 4) << 3);   // C layout: lanes>=16 hold M+8
#pragma unroll
  for (int r = 0; r < 8; ++r) {
    int row = rbase + r;
    if (row >= p.M) continue;
    float val = acc[r] + bv;
    if (p.relu && val < 0.f) val = 0.f;
    if (p.Cf)
      p.Cf[(long)(row / p.CSq) * p.Cbs + (long)(row % p.CSq) * p.ldcf + col] = val;
    if (p.Cb)
      p.Cb[(long)row * p.ldcb + col] = f2bf(val);
  }
}

// ---------------------------------------------------------------------------
// Multi-head attention core (dh = 192, NH = 4).  One block per (q-pos, head,
// batch).  Scores + softmax + weighted V sum; keys/values read as bf16.
// ---------------------------------------------------------------------------
__global__ void attention_kernel(const unsigned short* __restrict__ q, int ldq, int qoff,
                                 const unsigned short* __restrict__ k, int ldk, int koff,
                                 const unsigned short* __restrict__ v, int ldv, int voff,
                                 const int* __restrict__ mask,
                                 unsigned short* __restrict__ o,
                                 int Sq, int tk) {
  __shared__ float qs[192];
  __shared__ float sc[256];
  __shared__ float red[256];
  const int p = blockIdx.x, h = blockIdx.y, b = blockIdx.z;
  const int tid = threadIdx.x;

  long qrow = ((long)(b * Sq + p)) * ldq + qoff + h * 192;
  if (tid < 192) qs[tid] = bf2f(q[qrow + tid]);
  __syncthreads();

  float s = -3.4e38f;
  if (tid < tk) {
    const unsigned short* kr = k + ((long)(b * tk + tid)) * ldk + koff + h * 192;
    float acc = 0.f;
#pragma unroll 4
    for (int d = 0; d < 192; ++d) acc += qs[d] * bf2f(kr[d]);
    s = acc * 0.07216878364870323f;           // 1/sqrt(192)
    if (mask) s += (mask[b * tk + tid] == 0) ? -1e9f : 0.f;
  }
  red[tid] = s; __syncthreads();
  for (int w = 128; w > 0; w >>= 1) {
    if (tid < w) red[tid] = fmaxf(red[tid], red[tid + w]);
    __syncthreads();
  }
  const float mx = red[0]; __syncthreads();

  const float e = (tid < tk) ? __expf(s - mx) : 0.f;
  sc[tid] = e;
  red[tid] = e; __syncthreads();
  for (int w = 128; w > 0; w >>= 1) {
    if (tid < w) red[tid] += red[tid + w];
    __syncthreads();
  }
  const float inv = 1.f / red[0];

  if (tid < 192) {
    float acc = 0.f;
    for (int sdx = 0; sdx < tk; ++sdx)
      acc += sc[sdx] * bf2f(v[((long)(b * tk + sdx)) * ldv + voff + h * 192 + tid]);
    o[((long)(b * Sq + p)) * 768 + h * 192 + tid] = f2bf(acc * inv);
  }
}

// ---------------------------------------------------------------------------
// x = LayerNorm(xf + hf) * g + b ; writes fp32 (residual base) + bf16 (GEMM A)
// One block per row; 256 threads * 3 elements = 768.
// ---------------------------------------------------------------------------
__global__ void ln_kernel(float* __restrict__ xf, const float* __restrict__ hf,
                          const float* __restrict__ g, const float* __restrict__ bta,
                          unsigned short* __restrict__ xb) {
  __shared__ float red[256];
  const int m = blockIdx.x, tid = threadIdx.x;
  const long base = (long)m * 768;
  float vals[3]; float s = 0.f;
#pragma unroll
  for (int i = 0; i < 3; ++i) {
    int j = tid + i * 256;
    float t = xf[base + j] + hf[base + j];
    vals[i] = t; s += t;
  }
  red[tid] = s; __syncthreads();
  for (int w = 128; w > 0; w >>= 1) { if (tid < w) red[tid] += red[tid + w]; __syncthreads(); }
  const float mean = red[0] * (1.f / 768.f); __syncthreads();
  float s2 = 0.f;
#pragma unroll
  for (int i = 0; i < 3; ++i) { float d = vals[i] - mean; s2 += d * d; }
  red[tid] = s2; __syncthreads();
  for (int w = 128; w > 0; w >>= 1) { if (tid < w) red[tid] += red[tid + w]; __syncthreads(); }
  const float inv = rsqrtf(red[0] * (1.f / 768.f) + 1e-5f);
#pragma unroll
  for (int i = 0; i < 3; ++i) {
    int j = tid + i * 256;
    float y = (vals[i] - mean) * inv * g[j] + bta[j];
    xf[base + j] = y;
    xb[base + j] = f2bf(y);
  }
}

// ---------------------------------------------------------------------------
// dec[:,0,:] = template_emb[sentiments]
// ---------------------------------------------------------------------------
__global__ void init_dec_kernel(const float* __restrict__ templ, const int* __restrict__ sent,
                                float* __restrict__ dec_f, unsigned short* __restrict__ dec_b) {
  const int b = blockIdx.x;
  const float* src = templ + (long)sent[b] * 768;
  const long base = (long)b * 33 * 768;
  for (int j = threadIdx.x; j < 768; j += blockDim.x) {
    float val = src[j];
    dec_f[base + j] = val;
    dec_b[base + j] = f2bf(val);
  }
}

// ---------------------------------------------------------------------------
// argmax over V=32000 (first-max tie rule == jnp.argmax) + embedding gather
// emb = tok_w[:, id] + tok_b ; appended to dec as position t+1.
// ---------------------------------------------------------------------------
__global__ void argmax_embed_kernel(const float* __restrict__ logits, int t,
                                    const float* __restrict__ tok_w,
                                    const float* __restrict__ tok_b,
                                    float* __restrict__ dec_f,
                                    unsigned short* __restrict__ dec_b) {
  __shared__ float rv[256];
  __shared__ int ri[256];
  const int b = blockIdx.x, tid = threadIdx.x;
  const float* row = logits + ((long)b * 32 + t) * 32000;
  float best = -3.4e38f; int bi = 0x7fffffff;
  for (int j = tid; j < 32000; j += 256) {
    float val = row[j];
    if (val > best || (val == best && j < bi)) { best = val; bi = j; }
  }
  rv[tid] = best; ri[tid] = bi; __syncthreads();
  for (int w = 128; w > 0; w >>= 1) {
    if (tid < w) {
      float ov = rv[tid + w]; int oi = ri[tid + w];
      if (ov > rv[tid] || (ov == rv[tid] && oi < ri[tid])) { rv[tid] = ov; ri[tid] = oi; }
    }
    __syncthreads();
  }
  const int id = ri[0];
  const long base = ((long)b * 33 + (t + 1)) * 768;
  for (int j = tid; j < 768; j += 256) {
    float e = tok_w[(long)j * 32000 + id] + tok_b[j];
    dec_f[base + j] = e;
    dec_b[base + j] = f2bf(e);
  }
}

// ---------------------------------------------------------------------------
// Host orchestration
// ---------------------------------------------------------------------------
namespace {
constexpr int cB = 8, cS = 256, cH = 768, cH3 = 2304, cFF = 3072, cV = 32000;
constexpr int cNH = 4, cSTEPS = 32, cMAXPOS = 33, cKMAX = 3072;
}

extern "C" void kernel_launch(void* const* d_in, const int* in_sizes, int n_in,
                              void* d_out, int out_size, void* d_ws, size_t ws_size,
                              hipStream_t stream) {
  (void)in_sizes; (void)n_in; (void)out_size; (void)ws_size;
  const float* hidden      = (const float*)d_in[0];
  const int*   sent        = (const int*)d_in[1];
  const int*   amask       = (const int*)d_in[2];
  const float* templ       = (const float*)d_in[5];
  const float* ctx_in_w    = (const float*)d_in[6];
  const float* ctx_in_b    = (const float*)d_in[7];
  const float* ctx_out_w   = (const float*)d_in[8];
  const float* ctx_out_b   = (const float*)d_in[9];
  const float* self_in_w   = (const float*)d_in[10];
  const float* self_in_b   = (const float*)d_in[11];
  const float* self_out_w  = (const float*)d_in[12];
  const float* self_out_b  = (const float*)d_in[13];
  const float* cross_in_w  = (const float*)d_in[14];
  const float* cross_in_b  = (const float*)d_in[15];
  const float* cross_out_w = (const float*)d_in[16];
  const float* cross_out_b = (const float*)d_in[17];
  const float* ff1_w       = (const float*)d_in[18];
  const float* ff1_b       = (const float*)d_in[19];
  const float* ff2_w       = (const float*)d_in[20];
  const float* ff2_b       = (const float*)d_in[21];
  const float* norm_g      = (const float*)d_in[22];
  const float* norm_b      = (const float*)d_in[23];
  const float* out_w       = (const float*)d_in[24];
  const float* out_b       = (const float*)d_in[25];
  const float* tok_w       = (const float*)d_in[26];
  const float* tok_b       = (const float*)d_in[27];
  float* out = (float*)d_out;

  // --- workspace bump allocator (256B aligned) ------------------------------
  char* ws = (char*)d_ws; size_t off = 0;
  auto alloc = [&](size_t bytes) -> char* {
    char* ptr = ws + off; off = (off + bytes + 255) & ~(size_t)255; return ptr;
  };
  auto aB = [&](size_t e) -> unsigned short* { return (unsigned short*)alloc(e * 2); };
  auto aF = [&](size_t e) -> float* { return (float*)alloc(e * 4); };

  // bf16 weight copies (~92 MB total -> resident in the 192 MB L2)
  unsigned short* wb_ctx_in   = aB((size_t)cH3 * cH);
  unsigned short* wb_ctx_out  = aB((size_t)cH * cH);
  unsigned short* wb_self_in  = aB((size_t)2 * cH3 * cH);
  unsigned short* wb_self_out = aB((size_t)2 * cH * cH);
  unsigned short* wb_cross_in = aB((size_t)2 * cH3 * cH);
  unsigned short* wb_cross_out= aB((size_t)2 * cH * cH);
  unsigned short* wb_ff1      = aB((size_t)2 * cFF * cH);
  unsigned short* wb_ff2      = aB((size_t)2 * cH * cFF);
  unsigned short* wb_out      = aB((size_t)cV * cH);
  unsigned short* hs_b        = aB((size_t)cB * cS * cH);
  // step-invariant encoder K/V (ctx + 2 cross layers)
  unsigned short* Kc[3]; unsigned short* Vc[3];
  for (int a = 0; a < 3; ++a) { Kc[a] = aB((size_t)cB * cS * cH); Vc[a] = aB((size_t)cB * cS * cH); }
  // decoder sequence (persistent) and per-step activations
  float*          dec_f = aF((size_t)cB * cMAXPOS * cH);
  unsigned short* dec_b = aB((size_t)cB * cMAXPOS * cH);
  const int MMAX = cB * cSTEPS;                 // 256 rows max
  float*          xf   = aF((size_t)MMAX * cH);
  float*          hf   = aF((size_t)MMAX * cH);
  unsigned short* xb   = aB((size_t)MMAX * cH);
  unsigned short* qb   = aB((size_t)MMAX * cH);
  unsigned short* aob  = aB((size_t)MMAX * cH);
  unsigned short* hb   = aB((size_t)MMAX * cH);
  unsigned short* qkvb = aB((size_t)MMAX * cH3);
  unsigned short* m1b  = aB((size_t)MMAX * cFF);
  unsigned short* zrow = aB((size_t)cKMAX);     // zero dummy row for OOB lanes

  auto conv = [&](const float* src, unsigned short* dst, long n) {
    long blocks = (n + 255) / 256; if (blocks > 4096) blocks = 4096; if (blocks < 1) blocks = 1;
    f32_to_bf16_kernel<<<dim3((unsigned)blocks), dim3(256), 0, stream>>>(src, dst, n);
  };
  auto gemm = [&](const unsigned short* A, long lda, int ASq, long Abs,
                  const unsigned short* W, long ldw, const float* bias,
                  float* Cf, long ldcf, int CSq, long Cbs,
                  unsigned short* Cb, long ldcb,
                  int M, int N, int K, int relu) {
    GemmP pp{A, lda, ASq, Abs, W, ldw, bias, zrow, Cf, ldcf, CSq, Cbs, Cb, ldcb, M, N, K, relu};
    dim3 g((N + 127) / 128, (M + 15) / 16);
    gemm_bf16_wmma<<<g, dim3(256), 0, stream>>>(pp);
  };

  // --- one-time per call: zero dummy row, fp32 -> bf16 ----------------------
  zero_u16_kernel<<<dim3((cKMAX + 255) / 256), dim3(256), 0, stream>>>(zrow, cKMAX);
  conv(ctx_in_w,    wb_ctx_in,   (long)cH3 * cH);
  conv(ctx_out_w,   wb_ctx_out,  (long)cH * cH);
  conv(self_in_w,   wb_self_in,  2L * cH3 * cH);
  conv(self_out_w,  wb_self_out, 2L * cH * cH);
  conv(cross_in_w,  wb_cross_in, 2L * cH3 * cH);
  conv(cross_out_w, wb_cross_out,2L * cH * cH);
  conv(ff1_w,       wb_ff1,      2L * cFF * cH);
  conv(ff2_w,       wb_ff2,      2L * cH * cFF);
  conv(out_w,       wb_out,      (long)cV * cH);
  conv(hidden,      hs_b,        (long)cB * cS * cH);

  // --- one-time: encoder K/V projections (step-invariant) -------------------
  for (int a = 0; a < 3; ++a) {
    const unsigned short* W3 = (a == 0) ? wb_ctx_in : wb_cross_in + (size_t)(a - 1) * cH3 * cH;
    const float* b3 = (a == 0) ? ctx_in_b : cross_in_b + (a - 1) * cH3;
    gemm(hs_b, cH, cB * cS, 0, W3 + (size_t)cH * cH,  cH, b3 + cH,
         nullptr, 1, 1, 0, Kc[a], cH, cB * cS, cH, cH, 0);
    gemm(hs_b, cH, cB * cS, 0, W3 + (size_t)2 * cH * cH, cH, b3 + 2 * cH,
         nullptr, 1, 1, 0, Vc[a], cH, cB * cS, cH, cH, 0);
  }

  init_dec_kernel<<<dim3(cB), dim3(256), 0, stream>>>(templ, sent, dec_f, dec_b);

  // --- 32 sequential decode steps ------------------------------------------
  for (int t = 0; t < cSTEPS; ++t) {
    const int Sq = t + 1, M = cB * Sq;
    // context cross-attention: x = MHA(dec, hidden, hidden)
    gemm(dec_b, cH, Sq, (long)cMAXPOS * cH, wb_ctx_in, cH, ctx_in_b,
         nullptr, 1, 1, 0, qb, cH, M, cH, cH, 0);
    attention_kernel<<<dim3(Sq, cNH, cB), dim3(256), 0, stream>>>(
        qb, cH, 0, Kc[0], cH, 0, Vc[0], cH, 0, amask, aob, Sq, cS);
    gemm(aob, cH, M, 0, wb_ctx_out, cH, ctx_out_b,
         xf, cH, M, 0, xb, cH, M, cH, cH, 0);

    for (int i = 0; i < 2; ++i) {
      // self-attention
      gemm(xb, cH, M, 0, wb_self_in + (size_t)i * cH3 * cH, cH, self_in_b + i * cH3,
           nullptr, 1, 1, 0, qkvb, cH3, M, cH3, cH, 0);
      attention_kernel<<<dim3(Sq, cNH, cB), dim3(256), 0, stream>>>(
          qkvb, cH3, 0, qkvb, cH3, cH, qkvb, cH3, 2 * cH, nullptr, hb, Sq, Sq);
      gemm(hb, cH, M, 0, wb_self_out + (size_t)i * cH * cH, cH, self_out_b + i * cH,
           hf, cH, M, 0, nullptr, 1, M, cH, cH, 0);
      ln_kernel<<<dim3(M), dim3(256), 0, stream>>>(
          xf, hf, norm_g + (i * 3 + 0) * cH, norm_b + (i * 3 + 0) * cH, xb);
      // cross-attention over encoder states (uses precomputed K/V)
      gemm(xb, cH, M, 0, wb_cross_in + (size_t)i * cH3 * cH, cH, cross_in_b + i * cH3,
           nullptr, 1, 1, 0, qb, cH, M, cH, cH, 0);
      attention_kernel<<<dim3(Sq, cNH, cB), dim3(256), 0, stream>>>(
          qb, cH, 0, Kc[1 + i], cH, 0, Vc[1 + i], cH, 0, amask, hb, Sq, cS);
      gemm(hb, cH, M, 0, wb_cross_out + (size_t)i * cH * cH, cH, cross_out_b + i * cH,
           hf, cH, M, 0, nullptr, 1, M, cH, cH, 0);
      ln_kernel<<<dim3(M), dim3(256), 0, stream>>>(
          xf, hf, norm_g + (i * 3 + 1) * cH, norm_b + (i * 3 + 1) * cH, xb);
      // FFN (ReLU)
      gemm(xb, cH, M, 0, wb_ff1 + (size_t)i * cFF * cH, cH, ff1_b + i * cFF,
           nullptr, 1, 1, 0, m1b, cFF, M, cFF, cH, 1);
      gemm(m1b, cFF, M, 0, wb_ff2 + (size_t)i * cH * cFF, cFF, ff2_b + i * cH,
           hf, cH, M, 0, nullptr, 1, M, cH, cFF, 0);
      ln_kernel<<<dim3(M), dim3(256), 0, stream>>>(
          xf, hf, norm_g + (i * 3 + 2) * cH, norm_b + (i * 3 + 2) * cH, xb);
    }

    // logits of last position -> d_out[:, t, :]   (row b at (b*32 + t)*V)
    gemm(xb + (size_t)(Sq - 1) * cH, cH, 1, (long)Sq * cH,
         wb_out, cH, out_b,
         out + (size_t)t * cV, cV, 1, (long)cSTEPS * cV,
         nullptr, 1, cB, cV, cH, 0);
    // argmax + token embedding appended as decoder position t+1
    argmax_embed_kernel<<<dim3(cB), dim3(256), 0, stream>>>(
        out, t, tok_w, tok_b, dec_f, dec_b);
  }
}
